// CrossAttention_73486890435134
// MI455X (gfx1250) — matrix operand order repaired and checked by reading, not compiled, not run on hardware
//
#include <hip/hip_runtime.h>

// ---------------------------------------------------------------------------
// CDNA5 (gfx1250) cross-attention: QKV proj -> flash attention -> out proj
// Matmuls via v_wmma_f32_16x16x32_bf16 (f32 accumulate, wave32).
// Attention K/V tiles staged to LDS with async global->LDS (double-buffered).
// ---------------------------------------------------------------------------

typedef __attribute__((ext_vector_type(16))) __bf16 v16bf;
typedef __attribute__((ext_vector_type(8)))  float  v8f;
typedef __attribute__((ext_vector_type(4)))  unsigned int u32x4;
typedef __attribute__((ext_vector_type(4)))  int i32x4;

#if __has_builtin(__builtin_amdgcn_global_load_async_to_lds_b128) && \
    __has_builtin(__builtin_amdgcn_s_wait_asynccnt)
#define HAVE_ASYNC_LDS 1
#else
#define HAVE_ASYNC_LDS 0
#endif

union BF16V {
    v16bf        v;
    u32x4        q[2];
    unsigned int w[8];
    unsigned short h[16];
};

__device__ __forceinline__ unsigned short f2bf(float f) {
    unsigned u = __float_as_uint(f);
    u += 0x7fffu + ((u >> 16) & 1u);          // round-to-nearest-even
    return (unsigned short)(u >> 16);
}
__device__ __forceinline__ unsigned pk2(float a, float b) {
    return (unsigned)f2bf(a) | ((unsigned)f2bf(b) << 16);
}
__device__ __forceinline__ v8f wmma_bf16(v16bf a, v16bf b, v8f c) {
    // (neg_a, A, neg_b, B, c_mod, C, reuse_a, reuse_b)
    return __builtin_amdgcn_wmma_f32_16x16x32_bf16(false, a, false, b,
                                                   (short)0, c, false, false);
}
// A operand (16x32 bf16, M x K): lane = M row, slots[0..15] = K 16h+0..15
// (h = lane>>4). Caller passes p already offset by 16h.
__device__ __forceinline__ void load_a(BF16V& r, const unsigned short* p) {
    r.q[0] = *reinterpret_cast<const u32x4*>(p);
    r.q[1] = *reinterpret_cast<const u32x4*>(p + 8);
}
// B operand (32x16 bf16, K x N): lane = N col, slots[0..15] = K 16h+0..15.
// Caller passes p already offset by 16h.
__device__ __forceinline__ void load_b(BF16V& r, const unsigned short* p) {
    r.q[0] = *reinterpret_cast<const u32x4*>(p);
    r.q[1] = *reinterpret_cast<const u32x4*>(p + 8);
}
// 16-byte global -> LDS transfer (async on gfx1250 toolchains that expose it)
__device__ __forceinline__ void stage16(const unsigned short* g, unsigned short* l) {
#if HAVE_ASYNC_LDS
    __builtin_amdgcn_global_load_async_to_lds_b128(
        (__attribute__((address_space(1))) i32x4*)(g),
        (__attribute__((address_space(3))) i32x4*)(l), 0, 0);
#else
    *reinterpret_cast<u32x4*>(l) = *reinterpret_cast<const u32x4*>(g);
#endif
}
__device__ __forceinline__ void wait_async0() {
#if HAVE_ASYNC_LDS
    __builtin_amdgcn_s_wait_asynccnt(0);
#endif
}

// ---------------------------------------------------------------------------
// Kernel 0: one-shot f32 -> bf16 conversion (x, Wq, Wk, Wv, Wo)
// ---------------------------------------------------------------------------
__global__ __launch_bounds__(256) void cvt_kernel(
    const float* __restrict__ x,  const float* __restrict__ Wq,
    const float* __restrict__ Wk, const float* __restrict__ Wv,
    const float* __restrict__ Wo, unsigned short* __restrict__ dst) {
    const size_t NX = 4194304;                       // 2*4096*512
    size_t i = (size_t)blockIdx.x * 256 + threadIdx.x;
    float f;
    if (i < NX) {
        f = x[i];
    } else {
        size_t j = i - NX;                           // 4 x 262144 weights
        const float* tbl[4] = {Wq, Wk, Wv, Wo};
        f = tbl[j >> 18][j & 262143];
    }
    dst[i] = f2bf(f);
}

// ---------------------------------------------------------------------------
// Kernel 1: fused QKV projection.  One wave: 16 rows x 64 feats x {q,k,v}.
//   Qh [b,h,n,64] (pre-scaled by 0.125), Kh [b,h,n,64], VT [b,h,64,n]
// ---------------------------------------------------------------------------
__global__ __launch_bounds__(256) void qkv_kernel(
    const unsigned short* __restrict__ xb,
    const unsigned short* __restrict__ Wqb,
    const unsigned short* __restrict__ Wkb,
    const unsigned short* __restrict__ Wvb,
    unsigned short* __restrict__ Qh,
    unsigned short* __restrict__ Kh,
    unsigned short* __restrict__ VT) {
    const int lane = threadIdx.x & 31;
    const int half = lane >> 4, l15 = lane & 15;
    const int wave = blockIdx.x * 8 + (threadIdx.x >> 5);
    const int row0 = (wave >> 3) * 16;               // 512 row tiles
    const int og   = wave & 7;                       // head / 64-feature group
    const int o0   = og * 64;

    v8f acc[3][4] = {};
    const unsigned short* Ws[3] = {Wqb, Wkb, Wvb};
    const unsigned short* arow = xb + (size_t)(row0 + l15) * 512 + 16 * half;

    for (int k0 = 0; k0 < 512; k0 += 32) {
        BF16V A; load_a(A, arow + k0);
        #pragma unroll
        for (int w = 0; w < 3; ++w) {
            #pragma unroll
            for (int t = 0; t < 4; ++t) {
                BF16V Bv;
                load_b(Bv, Ws[w] + (size_t)(o0 + t * 16 + l15) * 512 + k0 + 16 * half);
                acc[w][t] = wmma_bf16(A.v, Bv.v, acc[w][t]);
            }
        }
    }

    const int b  = row0 >> 12;                       // N = 4096 rows per batch
    const int n0 = (row0 & 4095) + 8 * half;         // D rows: m = vg + 8h
    const int h  = og;
    #pragma unroll
    for (int t = 0; t < 4; ++t) {
        const int dd = t * 16 + l15;                 // D cols: n = l15
        const size_t qkbase = ((size_t)(b * 8 + h) * 4096) * 64 + dd;
        #pragma unroll
        for (int vg = 0; vg < 8; ++vg) {
            const size_t at = qkbase + (size_t)(n0 + vg) * 64;
            Qh[at] = f2bf(acc[0][t][vg] * 0.125f);   // fold softmax scale into Q
            Kh[at] = f2bf(acc[1][t][vg]);
        }
        u32x4 pkd;
        pkd[0] = pk2(acc[2][t][0], acc[2][t][1]);
        pkd[1] = pk2(acc[2][t][2], acc[2][t][3]);
        pkd[2] = pk2(acc[2][t][4], acc[2][t][5]);
        pkd[3] = pk2(acc[2][t][6], acc[2][t][7]);
        const size_t vtb = ((size_t)(b * 8 + h) * 64 + dd) * 4096 + n0;
        *reinterpret_cast<u32x4*>(VT + vtb) = pkd;
    }
}

// ---------------------------------------------------------------------------
// Kernel 2: flash attention.  4 waves/block share one (b,h); K/V blocks are
// double-buffered in LDS via async global->LDS.  One wave = 16 queries,
// 32 keys/iter.  S^T = K·Q^T (lane owns one query); O^T += V^T·P^T.
// ---------------------------------------------------------------------------
#define KSTR 72     // 64 + 8 pad (shorts); 144B row stride, 16B-multiple
#define VSTR 40     // 32 + 8 pad (shorts);  80B row stride, 16B-multiple

__global__ __launch_bounds__(128) void attn_kernel(
    const unsigned short* __restrict__ Qh,
    const unsigned short* __restrict__ Kh,
    const unsigned short* __restrict__ VT,
    unsigned short* __restrict__ O) {
    __shared__ unsigned short sK[2][32 * KSTR];
    __shared__ unsigned short sV[2][64 * VSTR];

    const int tid  = threadIdx.x;                    // 0..127
    const int lane = tid & 31;
    const int half = lane >> 4, l15 = lane & 15;
    const int wave = blockIdx.x * 4 + (tid >> 5);
    const int bh = wave >> 8;                        // same for all 4 waves
    const int q0 = (wave & 255) * 16;

    const unsigned short* Qb = Qh + (size_t)bh * 4096 * 64;
    const unsigned short* Kb = Kh + (size_t)bh * 4096 * 64;
    const unsigned short* Vb = VT + (size_t)bh * 64 * 4096;

    // stage one 32-key block (K: 32x64, V^T: 64x32) into LDS buffer `buf`
    auto stageKV = [&](int j, int buf) {
        #pragma unroll
        for (int s = 0; s < 2; ++s) {
            const int seg = tid + s * 128;           // 256 x 16B segments each
            const int krow = seg >> 3, kpart = seg & 7;
            stage16(Kb + (size_t)(j + krow) * 64 + kpart * 8,
                    &sK[buf][krow * KSTR + kpart * 8]);
            const int vrow = seg >> 2, vpart = seg & 3;
            stage16(Vb + (size_t)vrow * 4096 + j + vpart * 8,
                    &sV[buf][vrow * VSTR + vpart * 8]);
        }
    };

    BF16V bq0, bq1;                                  // Q^T B-operands, d 0..31 / 32..63
    {
        const unsigned short* qr = Qb + (size_t)(q0 + l15) * 64 + 16 * half;
        load_b(bq0, qr);
        load_b(bq1, qr + 32);
    }

    v8f acc[4] = {};                                 // O^T: 4 dim-tiles x 16 queries
    float m = -1e30f, l = 0.0f;
    const float L2E = 1.4426950408889634f;

    stageKV(0, 0);
    wait_async0();
    __syncthreads();

    for (int j0 = 0; j0 < 4096; j0 += 32) {
        const int cur = (j0 >> 5) & 1;
        if (j0 + 32 < 4096) stageKV(j0 + 32, cur ^ 1);   // async prefetch

        // ---- S^T tiles: keys 0..15 (s0) and 16..31 (s1) of this block ----
        v8f s0 = {}, s1 = {};
        {
            const unsigned short* kr0 = &sK[cur][0] + l15 * KSTR + 16 * half;
            const unsigned short* kr1 = kr0 + 16 * KSTR;
            BF16V a00, a01, a10, a11;
            load_a(a00, kr0); load_a(a01, kr0 + 32);
            load_a(a10, kr1); load_a(a11, kr1 + 32);
            s0 = wmma_bf16(a00.v, bq0.v, s0);
            s0 = wmma_bf16(a01.v, bq1.v, s0);
            s1 = wmma_bf16(a10.v, bq0.v, s1);
            s1 = wmma_bf16(a11.v, bq1.v, s1);
        }
        // ---- online softmax (lane owns query l15; halves synced via shfl) ----
        float mx = s0[0];
        #pragma unroll
        for (int i = 0; i < 8; ++i) { mx = fmaxf(mx, s0[i]); mx = fmaxf(mx, s1[i]); }
        mx = fmaxf(mx, __shfl_xor(mx, 16));
        const float mn  = fmaxf(m, mx);
        const float cor = __builtin_amdgcn_exp2f((m - mn) * L2E);
        float p0[8], p1[8], sum = 0.0f;
        #pragma unroll
        for (int i = 0; i < 8; ++i) {
            p0[i] = __builtin_amdgcn_exp2f((s0[i] - mn) * L2E);
            p1[i] = __builtin_amdgcn_exp2f((s1[i] - mn) * L2E);
            sum += p0[i] + p1[i];
        }
        sum += __shfl_xor(sum, 16);
        l = l * cor + sum;
        m = mn;
        #pragma unroll
        for (int t = 0; t < 4; ++t)
            #pragma unroll
            for (int i = 0; i < 8; ++i) acc[t][i] *= cor;

        // ---- build P^T B-operand: lane slots = keys 16h+0..15, query l15 ----
        // half0 lane holds keys 0..7 (p0) & 16..23 (p1); partner holds the rest.
        BF16V bp;
        #pragma unroll
        for (int i = 0; i < 4; ++i) {
            float sa = half ? p0[2 * i]     : p1[2 * i];      // send partner's need
            float sb = half ? p0[2 * i + 1] : p1[2 * i + 1];
            float ra = __shfl_xor(sa, 16);
            float rb = __shfl_xor(sb, 16);
            float lo0 = half ? ra : p0[2 * i];                // keys 16h+0..7
            float lo1 = half ? rb : p0[2 * i + 1];
            float hi0 = half ? p1[2 * i]     : ra;            // keys 16h+8..15
            float hi1 = half ? p1[2 * i + 1] : rb;
            bp.w[i]     = pk2(lo0, lo1);
            bp.w[4 + i] = pk2(hi0, hi1);
        }
        // ---- O^T += V^T_tile · P^T ----
        #pragma unroll
        for (int t = 0; t < 4; ++t) {
            const unsigned short* vr = &sV[cur][0] + (t * 16 + l15) * VSTR + 16 * half;
            BF16V av; load_a(av, vr);
            acc[t] = wmma_bf16(av.v, bp.v, acc[t]);
        }
        wait_async0();       // prefetch of next block landed
        __syncthreads();     // all waves done with `cur` before it is restaged
    }

    const float inv = 1.0f / l;
    const int bb = bh >> 3, hh = bh & 7;
    #pragma unroll
    for (int t = 0; t < 4; ++t) {
        u32x4 pkd;
        pkd[0] = pk2(acc[t][0] * inv, acc[t][1] * inv);
        pkd[1] = pk2(acc[t][2] * inv, acc[t][3] * inv);
        pkd[2] = pk2(acc[t][4] * inv, acc[t][5] * inv);
        pkd[3] = pk2(acc[t][6] * inv, acc[t][7] * inv);
        const size_t ob = ((size_t)(bb * 4096 + q0 + l15)) * 512
                        + hh * 64 + t * 16 + 8 * half;
        *reinterpret_cast<u32x4*>(O + ob) = pkd;
    }
}

// ---------------------------------------------------------------------------
// Kernel 3: output projection  out = O @ Wo^T + bo   (f32 output)
// ---------------------------------------------------------------------------
__global__ __launch_bounds__(256) void oproj_kernel(
    const unsigned short* __restrict__ O,
    const unsigned short* __restrict__ Wob,
    const float* __restrict__ bo,
    float* __restrict__ out) {
    const int lane = threadIdx.x & 31;
    const int half = lane >> 4, l15 = lane & 15;
    const int wave = blockIdx.x * 8 + (threadIdx.x >> 5);
    const int row0 = (wave >> 3) * 16;
    const int o0   = (wave & 7) * 64;

    v8f acc[4] = {};
    const unsigned short* arow = O + (size_t)(row0 + l15) * 512 + 16 * half;
    for (int k0 = 0; k0 < 512; k0 += 32) {
        BF16V A; load_a(A, arow + k0);
        #pragma unroll
        for (int t = 0; t < 4; ++t) {
            BF16V Bv;
            load_b(Bv, Wob + (size_t)(o0 + t * 16 + l15) * 512 + k0 + 16 * half);
            acc[t] = wmma_bf16(A.v, Bv.v, acc[t]);
        }
    }
    #pragma unroll
    for (int t = 0; t < 4; ++t) {
        const int o = o0 + t * 16 + l15;
        const float bias = bo[o];
        const int r0 = row0 + 8 * half;
        #pragma unroll
        for (int vg = 0; vg < 8; ++vg)
            out[(size_t)(r0 + vg) * 512 + o] = acc[t][vg] + bias;
    }
}

// ---------------------------------------------------------------------------
extern "C" void kernel_launch(void* const* d_in, const int* in_sizes, int n_in,
                              void* d_out, int out_size, void* d_ws, size_t ws_size,
                              hipStream_t stream) {
    (void)in_sizes; (void)n_in; (void)out_size; (void)ws_size;
    const float* x  = (const float*)d_in[0];
    const float* Wq = (const float*)d_in[1];
    const float* Wk = (const float*)d_in[2];
    const float* Wv = (const float*)d_in[3];
    const float* Wo = (const float*)d_in[4];
    const float* bo = (const float*)d_in[5];

    unsigned short* ws  = (unsigned short*)d_ws;
    unsigned short* xb  = ws;                    // 4,194,304 bf16
    unsigned short* Wqb = ws  + 4194304;         //   262,144
    unsigned short* Wkb = Wqb + 262144;
    unsigned short* Wvb = Wkb + 262144;
    unsigned short* Wob = Wvb + 262144;
    unsigned short* Qh  = Wob + 262144;          // 4,194,304
    unsigned short* Kh  = Qh  + 4194304;
    unsigned short* VT  = Kh  + 4194304;
    unsigned short* Obf = VT  + 4194304;         // total ~44 MB

    cvt_kernel  <<<20480, 256, 0, stream>>>(x, Wq, Wk, Wv, Wo, ws);
    qkv_kernel  <<<  512, 256, 0, stream>>>(xb, Wqb, Wkb, Wvb, Qh, Kh, VT);
    attn_kernel <<< 1024, 128, 0, stream>>>(Qh, Kh, VT, Obf);
    oproj_kernel<<<  512, 256, 0, stream>>>(Obf, Wob, bo, (float*)d_out);
}
